// DeformableTransformerDecoderLayer_17282948399177
// MI455X (gfx1250) — matrix operand order repaired and checked by reading, not compiled, not run on hardware
//
#include <hip/hip_runtime.h>
#include <hip/hip_bf16.h>

// ---------------------------------------------------------------------------
// DeformableTransformerDecoderLayer for MI455X (gfx1250, wave32).
//  - every GEMM runs on v_wmma_f32_16x16x32_bf16 (bf16 in, f32 accumulate)
//  - activations flow through workspace as bf16 (halves HBM traffic for the
//    bandwidth-bound parts; the roofline here is 23.3 TB/s, not FLOPs)
//  - GEMM A-tiles are fetched with the Tensor Data Mover (tensor_load_to_lds,
//    TENSORcnt) when the toolchain exposes the builtin
//  - butterfly reductions use ds_swizzle_b32 xor-masks (wave32 native)
// ---------------------------------------------------------------------------

typedef __attribute__((ext_vector_type(16))) __bf16 v16bf;
typedef __attribute__((ext_vector_type(8)))  float  v8f;
typedef __attribute__((ext_vector_type(4)))  unsigned int v4u;
typedef __attribute__((ext_vector_type(8)))  int v8i;
typedef __attribute__((ext_vector_type(4)))  int v4i;

union Frag { v16bf v; unsigned int u[8]; };

#if defined(__has_builtin)
#  if __has_builtin(__builtin_amdgcn_tensor_load_to_lds)
#    define HAVE_TDM 1
#  endif
#endif
#ifndef HAVE_TDM
#  define HAVE_TDM 0
#endif
#if defined(__has_include)
#  if __has_include(<hip/amd_detail/amd_gfx1250_TDM.h>)
#    define TDM_6ARG 1
#  endif
#endif
#ifndef TDM_6ARG
#  define TDM_6ARG 0
#endif

__device__ __forceinline__ unsigned short f32_to_bf16(float f) {
  unsigned int u = __float_as_uint(f);
  u += 0x7FFFu + ((u >> 16) & 1u);          // round-to-nearest-even
  return (unsigned short)(u >> 16);
}
__device__ __forceinline__ float bf16_to_f32(unsigned short h) {
  return __uint_as_float(((unsigned int)h) << 16);
}

// wave-local LDS store->load ordering (CDNA5 split counters)
#define WAVE_LDS_SYNC() asm volatile("s_wait_dscnt 0x0" ::: "memory")

// ds_swizzle xor-butterfly (group-of-32: offset = xor_mask<<10 | and_mask 0x1f)
#define SWZ_XOR(v, m) \
  __int_as_float(__builtin_amdgcn_ds_swizzle(__float_as_int(v), ((m) << 10) | 0x1f))

__device__ __forceinline__ float maxred16(float v) {
  v = fmaxf(v, SWZ_XOR(v, 8)); v = fmaxf(v, SWZ_XOR(v, 4));
  v = fmaxf(v, SWZ_XOR(v, 2)); v = fmaxf(v, SWZ_XOR(v, 1));
  return v;
}
__device__ __forceinline__ float sumred16(float v) {
  v += SWZ_XOR(v, 8); v += SWZ_XOR(v, 4);
  v += SWZ_XOR(v, 2); v += SWZ_XOR(v, 1);
  return v;
}
__device__ __forceinline__ float sumred32(float v) {
  v += SWZ_XOR(v, 16); v += SWZ_XOR(v, 8); v += SWZ_XOR(v, 4);
  v += SWZ_XOR(v, 2);  v += SWZ_XOR(v, 1);
  return v;
}

// ---------------------------------------------------------------------------
// Tiled GEMM:  Out[M,N] = act(A[M,K] @ B + bias)
//   ABF16=1 : A is bf16 (TDM tile DMA when available)
//   ABF16=0 : A is f32 (converted while staging; used for the `key` input)
//   WT=1    : W is (N,K) row-major  (x @ W^T)     WT=0 : W is (K,N)
// Block tile 128x64, 8 waves (4x2), 32x32 per wave, K step 32.
// ---------------------------------------------------------------------------
template <int ABF16, int WT, int RELU, int OUT_BF16, int MASK>
__global__ __launch_bounds__(256)
void gemm_bias_kernel(const void* __restrict__ Ap, const float* __restrict__ W,
                      const float* __restrict__ bias, void* __restrict__ OutP,
                      int M, int N, int K, const unsigned char* __restrict__ pad)
{
  constexpr int BM = 128, BN = 64, BK = 32;
  __shared__ unsigned short As[BM][BK];   // bf16 bits [m][k]
  __shared__ unsigned short Bs[BN][BK];   // bf16 bits transposed [n][k]

  const int tid  = threadIdx.x;
  const int wave = tid >> 5, lane = tid & 31;
  const int waveM = wave >> 1, waveN = wave & 1;
  const int l16 = lane & 15, lhi = lane >> 4;
  const int bm = blockIdx.x * BM, bn = blockIdx.y * BN;

  const float*          Af  = (const float*)Ap;
  const unsigned short* A16 = (const unsigned short*)Ap;

  v8f acc[2][2] = {};

  for (int kk = 0; kk < K; kk += BK) {
    // ---------------- stage A tile ----------------
    if (ABF16) {
#if HAVE_TDM
      if (wave == 0) {
        // Tensor DMA descriptor: 2D bf16 tile [128 x 32], row stride K elems.
        unsigned long long ga =
            (unsigned long long)(uintptr_t)(A16 + (size_t)bm * K + kk);
        unsigned int ldsAddr = (unsigned int)(uintptr_t)&As[0][0];
        int dim0 = K - kk;        // remaining elems in dim0 (OOB -> zero-fill)
        int dim1 = M - bm;        // remaining rows
        v4u g0;
        g0[0] = 1u;                                         // count=1
        g0[1] = ldsAddr;                                    // lds_addr
        g0[2] = (unsigned int)(ga & 0xffffffffull);         // global_addr lo
        g0[3] = (unsigned int)((ga >> 32) & 0x1ffffffull) | (2u << 30); // type=2
        v8i g1;
        g1[0] = (1 << 16);                                  // data_size=2B
        g1[1] = (dim0 & 0xffff) << 16;                      // tensor_dim0 lo16
        g1[2] = ((dim0 >> 16) & 0xffff) | ((dim1 & 0xffff) << 16);
        g1[3] = ((dim1 >> 16) & 0xffff) | (BK << 16);       // tile_dim0 = 32
        g1[4] = BM;                                         // tile_dim1, dim2=0
        g1[5] = K;                                          // dim0_stride lo32
        g1[6] = 0;
        g1[7] = 0;
        v4i g2 = {0, 0, 0, 0}, g3 = {0, 0, 0, 0};
#if TDM_6ARG
        v8i g4 = {0, 0, 0, 0, 0, 0, 0, 0};
        __builtin_amdgcn_tensor_load_to_lds(g0, g1, g2, g3, g4, 0);
#else
        __builtin_amdgcn_tensor_load_to_lds(g0, g1, g2, g3, 0);
#endif
      }
#else
      // fallback: plain coalesced dword copy of the bf16 tile
      unsigned int* As32 = (unsigned int*)&As[0][0];
      const unsigned int* Ag = (const unsigned int*)A16;
#pragma unroll
      for (int i = 0; i < 8; ++i) {
        int idx = tid + i * 256;          // 2048 dwords
        int r = idx >> 4, cu = idx & 15;
        int gr = bm + r;
        As32[idx] = (gr < M) ? Ag[(size_t)gr * (K >> 1) + (kk >> 1) + cu] : 0u;
      }
#endif
    } else {
      // f32 input (the `key` tensor): convert while staging
#pragma unroll
      for (int i = 0; i < 16; ++i) {
        int idx = tid + i * 256;
        int r = idx >> 5, c = idx & 31;
        int gr = bm + r;
        float val = (gr < M) ? Af[(size_t)gr * K + kk + c] : 0.f;
        As[r][c] = f32_to_bf16(val);
      }
    }
    // ---------------- stage B tile (transposed into LDS) ----------------
#pragma unroll
    for (int i = 0; i < 8; ++i) {
      int idx = tid + i * 256;
      if (WT) {                       // W is (N,K): direct rows
        int r = idx >> 5, c = idx & 31;
        Bs[r][c] = f32_to_bf16(W[(size_t)(bn + r) * K + kk + c]);
      } else {                        // W is (K,N): transpose while staging
        int n = idx & 63, k = idx >> 6;
        Bs[n][k] = f32_to_bf16(W[(size_t)(kk + k) * N + bn + n]);
      }
    }
#if HAVE_TDM
    if (ABF16 && wave == 0) __builtin_amdgcn_s_wait_tensorcnt(0);
#endif
    __syncthreads();

    // ---------------- fragments + WMMA ----------------
    Frag a[2], b[2];
#pragma unroll
    for (int s = 0; s < 2; ++s) {
      int mrow = waveM * 32 + s * 16 + l16;
      int ncol = waveN * 32 + s * 16 + l16;
#pragma unroll
      for (int r = 0; r < 8; ++r) {
        int ka = ((r >> 2) << 4) + (lhi << 3) + ((r & 3) << 1);
        int kb = ((r >> 2) << 4) + (lhi << 3) + ((r & 3) << 1);
        a[s].u[r] = *(const unsigned int*)&As[mrow][ka];
        b[s].u[r] = *(const unsigned int*)&Bs[ncol][kb];
      }
    }
#pragma unroll
    for (int sm = 0; sm < 2; ++sm)
#pragma unroll
      for (int sn = 0; sn < 2; ++sn)
        acc[sm][sn] = __builtin_amdgcn_wmma_f32_16x16x32_bf16(
            false, a[sm].v, false, b[sn].v, (short)0, acc[sm][sn],
            false, false);
    __syncthreads();
  }

  // ---------------- epilogue ----------------
  float* Of = (float*)OutP;
  unsigned short* Oh = (unsigned short*)OutP;
#pragma unroll
  for (int sm = 0; sm < 2; ++sm) {
#pragma unroll
    for (int sn = 0; sn < 2; ++sn) {
      int gcol = bn + waveN * 32 + sn * 16 + l16;
      float bv = bias[gcol];
#pragma unroll
      for (int r = 0; r < 8; ++r) {
        int grow = bm + waveM * 32 + sm * 16 + (lhi << 3) + r;
        if (grow < M) {
          float v = acc[sm][sn][r] + bv;
          if (RELU) v = v > 0.f ? v : 0.f;
          if (MASK) {                 // rows are (pix*BS + b); pad[b, pix]
            int b_ = grow & 7, l_ = grow >> 3;
            if (pad[(size_t)b_ * 17821 + l_]) v = 0.f;
          }
          if (OUT_BF16) Oh[(size_t)grow * N + gcol] = f32_to_bf16(v);
          else          Of[(size_t)grow * N + gcol] = v;
        }
      }
    }
  }
}

// ---------------------------------------------------------------------------
// Single-wave flash attention: one wave handles 16 queries for one (b,h).
// grid = (57, 64), block = 32.  Output bf16 (feeds the W_o WMMA GEMM).
// ---------------------------------------------------------------------------
__global__ __launch_bounds__(32)
void flash_attn_kernel(const unsigned short* __restrict__ qh,
                       const unsigned short* __restrict__ kh,
                       const unsigned short* __restrict__ vh,
                       unsigned short* __restrict__ sa)
{
  constexpr int LQ = 900;
  __shared__ unsigned short Qs[16][32];
  __shared__ unsigned short Ks[32][32];   // [key][d]
  __shared__ unsigned short Vs[32][32];   // [d][key]  (Bt layout)
  __shared__ unsigned short Ps[16][32];   // probs re-fragmented as A

  const int lane = threadIdx.x;
  const int l16 = lane & 15, lhi = lane >> 4;
  const int bh = blockIdx.y, h = bh & 7, b = bh >> 3;
  const int q0 = blockIdx.x * 16;

  {
    int q = q0 + l16;
    const unsigned short* qp = &qh[((size_t)q * 8 + b) * 256 + h * 32];
    for (int d = lhi * 16; d < lhi * 16 + 16; ++d)
      Qs[l16][d] = (q < LQ) ? qp[d] : (unsigned short)0;
  }
  WAVE_LDS_SYNC();
  Frag aq;
#pragma unroll
  for (int r = 0; r < 8; ++r) {
    int k = ((r >> 2) << 4) + (lhi << 3) + ((r & 3) << 1);
    aq.u[r] = *(const unsigned int*)&Qs[l16][k];
  }

  v8f o0 = {}, o1 = {};
  float mrow[8], lrow[8];
#pragma unroll
  for (int r = 0; r < 8; ++r) { mrow[r] = -1e30f; lrow[r] = 0.f; }

  const float scale = 0.17677669529663687f;   // 1/sqrt(32)

  for (int kc = 0; kc < LQ; kc += 32) {
    {
      int kg = kc + lane;
      bool ok = kg < LQ;
      const unsigned short* kp = &kh[((size_t)kg * 8 + b) * 256 + h * 32];
      const unsigned short* vp = &vh[((size_t)kg * 8 + b) * 256 + h * 32];
      for (int d = 0; d < 32; ++d) {
        Ks[lane][d] = ok ? kp[d] : (unsigned short)0;
        Vs[d][lane] = ok ? vp[d] : (unsigned short)0;
      }
    }
    WAVE_LDS_SYNC();
    Frag bk0, bk1, bv0, bv1;
#pragma unroll
    for (int r = 0; r < 8; ++r) {
      int k = ((r >> 2) << 4) + (lhi << 3) + ((r & 3) << 1);
      bk0.u[r] = *(const unsigned int*)&Ks[l16][k];
      bk1.u[r] = *(const unsigned int*)&Ks[16 + l16][k];
      bv0.u[r] = *(const unsigned int*)&Vs[l16][k];
      bv1.u[r] = *(const unsigned int*)&Vs[16 + l16][k];
    }
    v8f z = {};
    v8f s0 = __builtin_amdgcn_wmma_f32_16x16x32_bf16(false, aq.v, false, bk0.v,
                                                     (short)0, z, false, false);
    v8f s1 = __builtin_amdgcn_wmma_f32_16x16x32_bf16(false, aq.v, false, bk1.v,
                                                     (short)0, z, false, false);
    bool pad0 = (kc + l16) >= LQ;
    bool pad1 = (kc + 16 + l16) >= LQ;
#pragma unroll
    for (int r = 0; r < 8; ++r) {
      float a0 = pad0 ? -1e30f : s0[r] * scale;
      float a1 = pad1 ? -1e30f : s1[r] * scale;
      float mx = maxred16(fmaxf(a0, a1));
      float newm = fmaxf(mrow[r], mx);
      float p0 = __expf(a0 - newm), p1 = __expf(a1 - newm);
      float rs = sumred16(p0 + p1);
      float cf = __expf(mrow[r] - newm);
      mrow[r] = newm;
      lrow[r] = lrow[r] * cf + rs;
      o0[r] *= cf;
      o1[r] *= cf;
      Ps[(lhi << 3) + r][l16]      = f32_to_bf16(p0);
      Ps[(lhi << 3) + r][16 + l16] = f32_to_bf16(p1);
    }
    WAVE_LDS_SYNC();
    Frag ap;
#pragma unroll
    for (int r = 0; r < 8; ++r) {
      int k = ((r >> 2) << 4) + (lhi << 3) + ((r & 3) << 1);
      ap.u[r] = *(const unsigned int*)&Ps[l16][k];
    }
    o0 = __builtin_amdgcn_wmma_f32_16x16x32_bf16(false, ap.v, false, bv0.v,
                                                 (short)0, o0, false, false);
    o1 = __builtin_amdgcn_wmma_f32_16x16x32_bf16(false, ap.v, false, bv1.v,
                                                 (short)0, o1, false, false);
    WAVE_LDS_SYNC();
  }

#pragma unroll
  for (int r = 0; r < 8; ++r) {
    int q = q0 + (lhi << 3) + r;
    if (q < LQ) {
      float inv = 1.0f / lrow[r];
      size_t base = ((size_t)q * 8 + b) * 256 + h * 32;
      sa[base + l16]      = f32_to_bf16(o0[r] * inv);
      sa[base + 16 + l16] = f32_to_bf16(o1[r] * inv);
    }
  }
}

// ---------------------------------------------------------------------------
// MSDeformAttn sampling: one wave per (q,b,h); lane = channel (DH=32).
// Gathers 64B coalesced rows from the bf16 value tensor; writes bf16.
// ---------------------------------------------------------------------------
__global__ __launch_bounds__(256)
void msda_sample_kernel(const float* __restrict__ off,
                        const float* __restrict__ attnw,
                        const unsigned short* __restrict__ v,
                        const float* __restrict__ refp,
                        unsigned short* __restrict__ out)
{
  const int shapesH[4] = {100, 50, 25, 13};
  const int shapesW[4] = {134, 67, 34, 17};
  const int lstart[4]  = {0, 13400, 16750, 17600};

  int wave = threadIdx.x >> 5, lane = threadIdx.x & 31;
  int t = blockIdx.x * 8 + wave;
  int h = t & 7, b = (t >> 3) & 7, q = t >> 6;
  if (q >= 900) return;
  size_t rowq = (size_t)q * 8 + b;

  const float* aw = &attnw[rowq * 128 + h * 16];
  float w[16], mx = -1e30f;
#pragma unroll
  for (int i = 0; i < 16; ++i) { w[i] = aw[i]; mx = fmaxf(mx, w[i]); }
  float ssum = 0.f;
#pragma unroll
  for (int i = 0; i < 16; ++i) { w[i] = __expf(w[i] - mx); ssum += w[i]; }
  float inv = 1.f / ssum;

  const float* ob = &off[rowq * 256 + h * 32];
  float acc = 0.f;
#pragma unroll
  for (int l = 0; l < 4; ++l) {
    const int Hl = shapesH[l], Wl = shapesW[l];
    float rx = refp[(rowq * 4 + l) * 2 + 0];
    float ry = refp[(rowq * 4 + l) * 2 + 1];
    const unsigned short* vbase =
        v + ((size_t)lstart[l] * 8 + b) * 256 + h * 32 + lane;
#pragma unroll
    for (int p = 0; p < 4; ++p) {
      float ox = ob[l * 8 + p * 2 + 0];
      float oy = ob[l * 8 + p * 2 + 1];
      float x = (rx + ox / (float)Wl) * (float)Wl - 0.5f;
      float y = (ry + oy / (float)Hl) * (float)Hl - 0.5f;
      float x0f = floorf(x), y0f = floorf(y);
      int x0 = (int)x0f, y0 = (int)y0f;
      float fx = x - x0f, fy = y - y0f;
      float wgt = w[l * 4 + p] * inv;
      float sample = 0.f;
#pragma unroll
      for (int tap = 0; tap < 4; ++tap) {
        int dx = tap & 1, dy = tap >> 1;
        int xi = x0 + dx, yi = y0 + dy;
        float tw = (dx ? fx : 1.f - fx) * (dy ? fy : 1.f - fy);
        if (xi >= 0 && xi < Wl && yi >= 0 && yi < Hl) {
          size_t idx = (size_t)(yi * Wl + xi) * 2048;   // 8*256 elems/pixel
          sample += tw * bf16_to_f32(vbase[idx]);
        }
      }
      acc += wgt * sample;
    }
  }
  out[rowq * 256 + h * 32 + lane] = f32_to_bf16(acc);
}

// ---------------------------------------------------------------------------
// Fused residual + LayerNorm (D=256): one wave per row.
// Optional bf16 mirror of the LN result and of (LN + extra).
// ---------------------------------------------------------------------------
__global__ __launch_bounds__(256)
void ln_fused_kernel(const float* __restrict__ a, const float* __restrict__ bres,
                     const float* __restrict__ g, const float* __restrict__ beta,
                     float* __restrict__ out,
                     unsigned short* __restrict__ out_bf,
                     const float* __restrict__ extra,
                     unsigned short* __restrict__ out2_bf, int M)
{
  int wave = threadIdx.x >> 5, lane = threadIdx.x & 31;
  int row = blockIdx.x * 8 + wave;
  if (row >= M) return;
  size_t base = (size_t)row * 256;
  float x[8], sum = 0.f;
#pragma unroll
  for (int i = 0; i < 8; ++i) {
    int c = lane + i * 32;
    x[i] = a[base + c] + bres[base + c];
    sum += x[i];
  }
  sum = sumred32(sum);
  float mean = sum * (1.f / 256.f);
  float var = 0.f;
#pragma unroll
  for (int i = 0; i < 8; ++i) { float d = x[i] - mean; var += d * d; }
  var = sumred32(var);
  float rstd = rsqrtf(var * (1.f / 256.f) + 1e-5f);
#pragma unroll
  for (int i = 0; i < 8; ++i) {
    int c = lane + i * 32;
    float yv = (x[i] - mean) * rstd * g[c] + beta[c];
    out[base + c] = yv;
    if (out_bf)  out_bf[base + c]  = f32_to_bf16(yv);
    if (out2_bf) out2_bf[base + c] = f32_to_bf16(yv + extra[base + c]);
  }
}

__global__ __launch_bounds__(256)
void add_bf16_kernel(const float* __restrict__ a, const float* __restrict__ b,
                     unsigned short* __restrict__ o, int n)
{
  int i = blockIdx.x * 256 + threadIdx.x;
  if (i < n) o[i] = f32_to_bf16(a[i] + b[i]);
}

__global__ __launch_bounds__(256)
void cvt_bf16_kernel(const float* __restrict__ a, unsigned short* __restrict__ o,
                     int n)
{
  int i = blockIdx.x * 256 + threadIdx.x;
  if (i < n) o[i] = f32_to_bf16(a[i]);
}

// ---------------------------------------------------------------------------
extern "C" void kernel_launch(void* const* d_in, const int* in_sizes, int n_in,
                              void* d_out, int out_size, void* d_ws, size_t ws_size,
                              hipStream_t stream)
{
  (void)in_sizes; (void)n_in; (void)out_size; (void)ws_size;
  const float* query     = (const float*)d_in[0];
  const float* query_pos = (const float*)d_in[1];
  const float* key       = (const float*)d_in[2];
  const float* refp      = (const float*)d_in[3];
  const unsigned char* pad = (const unsigned char*)d_in[6];
  const float* W_off  = (const float*)d_in[7];
  const float* b_off  = (const float*)d_in[8];
  const float* W_attn = (const float*)d_in[9];
  const float* b_attn = (const float*)d_in[10];
  const float* W_val  = (const float*)d_in[11];
  const float* b_val  = (const float*)d_in[12];
  const float* W_out  = (const float*)d_in[13];
  const float* b_out  = (const float*)d_in[14];
  const float* W_qkv  = (const float*)d_in[15];
  const float* b_qkv  = (const float*)d_in[16];
  const float* W_o    = (const float*)d_in[17];
  const float* b_o    = (const float*)d_in[18];
  const float* g1     = (const float*)d_in[19];
  const float* beta1  = (const float*)d_in[20];
  const float* g2     = (const float*)d_in[21];
  const float* beta2  = (const float*)d_in[22];
  const float* g3     = (const float*)d_in[23];
  const float* beta3  = (const float*)d_in[24];
  const float* W1     = (const float*)d_in[25];
  const float* b1     = (const float*)d_in[26];
  const float* W2     = (const float*)d_in[27];
  const float* b2     = (const float*)d_in[28];

  // --------- workspace layout (bytes); buffers reused after last use -------
  char* ws = (char*)d_ws;
  unsigned short* qk_bf    = (unsigned short*)(ws + 0);          // 3.69 MB
  unsigned short* query_bf = (unsigned short*)(ws +  3686400);
  unsigned short* qh       = (unsigned short*)(ws +  7372800);
  unsigned short* kh       = (unsigned short*)(ws + 11059200);
  unsigned short* vhh      = (unsigned short*)(ws + 14745600);
  unsigned short* sa_bf    = (unsigned short*)(ws + 18432000);
  float*          saP      = (float*)(ws + 22118400);            // 7.37 MB
  float*          x        = (float*)(ws + 29491200);
  unsigned short* x2p_bf   = (unsigned short*)(ws + 36864000);
  float*          offB     = (float*)(ws + 40550400);
  float*          attnw    = (float*)(ws + 47923200);
  unsigned short* vval     = (unsigned short*)(ws + 51609600);   // 73.0 MB
  unsigned short* msda_bf  = (unsigned short*)(ws + 0);          // reuse qk_bf
  float*          ca       = (float*)(ws + 22118400);            // reuse saP
  float*          x2       = (float*)(ws + 40550400);            // reuse offB
  unsigned short* x2_bf    = (unsigned short*)(ws +  3686400);   // reuse query_bf
  unsigned short* ffn1_bf  = (unsigned short*)(ws +  7372800);   // reuse qh..sa_bf
  float*          ffn2     = (float*)(ws + 47923200);            // reuse attnw+vval
  // peak requirement ~125 MB

  const int Mq = 7200;      // NQ*BS
  const int Mv = 142568;    // LV*BS

  // 1) bf16 producers for the WMMA GEMM inputs
  add_bf16_kernel<<<7200, 256, 0, stream>>>(query, query_pos, qk_bf, 1843200);
  cvt_bf16_kernel<<<7200, 256, 0, stream>>>(query, query_bf, 1843200);

  // 2-4) QKV head projections (x @ W^T form), bf16 out
  gemm_bias_kernel<1,1,0,1,0><<<dim3(57,4), 256, 0, stream>>>(
      qk_bf,    W_qkv,          b_qkv,       qh,  Mq, 256, 256, nullptr);
  gemm_bias_kernel<1,1,0,1,0><<<dim3(57,4), 256, 0, stream>>>(
      qk_bf,    W_qkv + 65536,  b_qkv + 256, kh,  Mq, 256, 256, nullptr);
  gemm_bias_kernel<1,1,0,1,0><<<dim3(57,4), 256, 0, stream>>>(
      query_bf, W_qkv + 131072, b_qkv + 512, vhh, Mq, 256, 256, nullptr);

  // 5) self-attention (flash, WMMA)
  flash_attn_kernel<<<dim3(57,64), 32, 0, stream>>>(qh, kh, vhh, sa_bf);

  // 6) output proj (x @ W_o^T)
  gemm_bias_kernel<1,1,0,0,0><<<dim3(57,4), 256, 0, stream>>>(
      sa_bf, W_o, b_o, saP, Mq, 256, 256, nullptr);

  // 7) x = LN(query + sa); x2p_bf = bf16(x + query_pos)
  ln_fused_kernel<<<900, 256, 0, stream>>>(query, saP, g2, beta2, x,
                                           nullptr, query_pos, x2p_bf, Mq);

  // 8) value projection (142568x256x256), f32 A, masked, bf16 out
  gemm_bias_kernel<0,0,0,1,1><<<dim3(1114,4), 256, 0, stream>>>(
      key, W_val, b_val, vval, Mv, 256, 256, pad);

  // 9-10) sampling offsets + attention logits
  gemm_bias_kernel<1,0,0,0,0><<<dim3(57,4), 256, 0, stream>>>(
      x2p_bf, W_off, b_off, offB, Mq, 256, 256, nullptr);
  gemm_bias_kernel<1,0,0,0,0><<<dim3(57,2), 256, 0, stream>>>(
      x2p_bf, W_attn, b_attn, attnw, Mq, 128, 256, nullptr);

  // 11) deformable bilinear sampling
  msda_sample_kernel<<<7200, 256, 0, stream>>>(offB, attnw, vval, refp, msda_bf);

  // 12) cross-attn output proj
  gemm_bias_kernel<1,0,0,0,0><<<dim3(57,4), 256, 0, stream>>>(
      msda_bf, W_out, b_out, ca, Mq, 256, 256, nullptr);

  // 13) x2 = LN(x + ca), plus bf16 mirror for the FFN
  ln_fused_kernel<<<900, 256, 0, stream>>>(x, ca, g1, beta1, x2,
                                           x2_bf, nullptr, nullptr, Mq);

  // 14-15) FFN 256 -> 1024 (ReLU, bf16) -> 256
  gemm_bias_kernel<1,0,1,1,0><<<dim3(57,16), 256, 0, stream>>>(
      x2_bf, W1, b1, ffn1_bf, Mq, 1024, 256, nullptr);
  gemm_bias_kernel<1,0,0,0,0><<<dim3(57,4), 256, 0, stream>>>(
      ffn1_bf, W2, b2, ffn2, Mq, 256, 1024, nullptr);

  // 16) final LN -> d_out
  ln_fused_kernel<<<900, 256, 0, stream>>>(x2, ffn2, g3, beta3,
                                           (float*)d_out, nullptr, nullptr,
                                           nullptr, Mq);
}
